// ContrastiveLoss_45303315038970
// MI455X (gfx1250) — compile-verified, matrix operand order checked by reading, not run on hardware
//
#include <hip/hip_runtime.h>
#include <math.h>

// Problem constants (from reference setup_inputs)
#define NB 128      // batch (images == captions)
#define NR 36       // regions per image
#define NRP 48      // regions padded to 3x16 for GEMM1 M-tiles
#define NRP2 64     // regions padded to 2x32 for GEMM2 K
#define NW 64       // words per caption
#define ND 1024     // feature dim

typedef __attribute__((ext_vector_type(8)))  float  v8f;
typedef __attribute__((ext_vector_type(16))) __bf16 v16bf;
typedef __attribute__((ext_vector_type(8)))  __bf16 v8bf;

// ---------------- helpers ----------------

__device__ __forceinline__ unsigned short f2bf(float f) {
    union { float f; unsigned u; } x; x.f = f;
    unsigned u = x.u;
    unsigned r = u + 0x7FFFu + ((u >> 16) & 1u);   // round-to-nearest-even
    return (unsigned short)(r >> 16);
}

// Load one 16x16x32 bf16 operand fragment for this lane.
// p = row base + kbase + (lane>=16 ? 8 : 0). Layout: elems 0..7 = K{+0..+7},
// elems 8..15 = K{+16..+23} (hi lanes get +8 automatically via p).
__device__ __forceinline__ v16bf load_frag(const unsigned short* p) {
    v16bf r;
    ((v8bf*)&r)[0] = ((const v8bf*)(p))[0];
    ((v8bf*)&r)[1] = ((const v8bf*)(p + 16))[0];
    return r;
}

__device__ __forceinline__ v8f wmma_bf16(v16bf a, v16bf b, v8f c) {
    return __builtin_amdgcn_wmma_f32_16x16x32_bf16(false, a, false, b,
                                                   (short)0, c, false, false);
}

// ---------------- prepass: im -> bf16 (row-major, padded) + bf16 transposed ----------------

__global__ void k_prep_im(const float* __restrict__ im,
                          unsigned short* __restrict__ im_bf,     // [NB][NRP][ND]
                          unsigned short* __restrict__ im_t_bf) { // [NB][ND][NRP2]
    const int i = blockIdx.x;
    const float* src = im + (size_t)i * NR * ND;
    unsigned short* d0 = im_bf  + (size_t)i * NRP * ND;
    unsigned short* d1 = im_t_bf + (size_t)i * ND * NRP2;
    for (int idx = threadIdx.x; idx < NRP * ND; idx += blockDim.x) {
        int r = idx >> 10, d = idx & (ND - 1);
        d0[idx] = (r < NR) ? f2bf(src[r * ND + d]) : (unsigned short)0;
    }
    for (int idx = threadIdx.x; idx < ND * NRP2; idx += blockDim.x) {
        int d = idx >> 6, r = idx & (NRP2 - 1);
        d1[idx] = (r < NR) ? f2bf(src[r * ND + d]) : (unsigned short)0;
    }
}

// ---------------- prepass: s -> bf16 + caption word norms ----------------

__global__ void k_prep_s(const float* __restrict__ s,
                         unsigned short* __restrict__ s_bf,  // [NB][NW][ND]
                         float* __restrict__ ncap) {         // [NB][NW]
    const int j = blockIdx.x;
    const float* src = s + (size_t)j * NW * ND;
    unsigned short* d0 = s_bf + (size_t)j * NW * ND;
    for (int idx = threadIdx.x; idx < NW * ND; idx += blockDim.x)
        d0[idx] = f2bf(src[idx]);
    if (threadIdx.x < NW) {
        int w = threadIdx.x;
        float ss = 0.f;
        for (int d = 0; d < ND; ++d) { float v = src[w * ND + d]; ss += v * v; }
        ncap[(size_t)j * NW + w] = sqrtf(ss);
    }
}

// ---------------- main fused pair kernel: one block per (image i, caption j) ----------------

#define ASTR 68   // fp32 attn LDS row stride (floats)
#define TSTR 72   // bf16 attn_t LDS row stride (ushorts), 16B-aligned

__global__ __launch_bounds__(256) void k_pair(
        const unsigned short* __restrict__ im_bf,    // [NB][NRP][ND]
        const unsigned short* __restrict__ im_t_bf,  // [NB][ND][NRP2]
        const unsigned short* __restrict__ s_bf,     // [NB][NW][ND]
        const float* __restrict__ s_f32,             // original captions
        const int* __restrict__ s_l,
        const float* __restrict__ ncap,              // [NB][NW]
        float* __restrict__ scores) {                // [NB][NB]
    __shared__ float          attnS[NRP * ASTR];     // fp32 attn [r][w]
    __shared__ unsigned short attnT[NW * TSTR];      // bf16 softmax(attn)^T [w][r]
    __shared__ float          numP[2 * NW];          // partial num per d-half
    __shared__ float          sqP[2 * NW];           // partial |wei|^2 per d-half
    __shared__ float          wred[8];

    const int i = blockIdx.x;   // image
    const int j = blockIdx.y;   // caption
    const int lane = threadIdx.x & 31;
    const int wv   = threadIdx.x >> 5;
    const int ln16 = lane & 15;
    const int hi   = lane >> 4;
    const int sel  = hi ? 8 : 0;
    const int slen = s_l[j];

    // ---- GEMM1: attn[r][w] = im[i] @ cap[j]^T, bf16 WMMA, K=1024 ----
    const unsigned short* Aim = im_bf + (size_t)i * NRP * ND;
    const unsigned short* Bcp = s_bf  + (size_t)j * NW  * ND;
    for (int t = wv; t < 12; t += 8) {              // 3 M-tiles x 4 N-tiles
        const int mt = t >> 2, nt = t & 3;
        v8f acc = {0.f, 0.f, 0.f, 0.f, 0.f, 0.f, 0.f, 0.f};
        const unsigned short* ar = Aim + (size_t)(mt * 16 + ln16) * ND + sel;
        const unsigned short* br = Bcp + (size_t)(nt * 16 + ln16) * ND + sel;
#pragma unroll 4
        for (int ks = 0; ks < ND / 32; ++ks) {
            v16bf a = load_frag(ar + ks * 32);
            v16bf b = load_frag(br + ks * 32);
            acc = wmma_bf16(a, b, acc);
        }
        const int w = nt * 16 + ln16;
#pragma unroll
        for (int v = 0; v < 8; ++v) {
            const int r = mt * 16 + v + (hi ? 8 : 0);
            attnS[r * ASTR + w] = acc[v];
        }
    }
    __syncthreads();

    // ---- leaky_relu(0.1) * word-mask, L2-normalize over words per region ----
    for (int r = wv; r < NR; r += 8) {
        float v0 = attnS[r * ASTR + lane];
        float v1 = attnS[r * ASTR + lane + 32];
        v0 = v0 > 0.f ? v0 : 0.1f * v0;
        v1 = v1 > 0.f ? v1 : 0.1f * v1;
        if (lane >= slen)      v0 = 0.f;
        if (lane + 32 >= slen) v1 = 0.f;
        float ss = v0 * v0 + v1 * v1;
        for (int m = 16; m; m >>= 1) ss += __shfl_xor(ss, m, 32);
        const float inv = 1.f / (sqrtf(ss) + 1e-8f);
        attnS[r * ASTR + lane]      = v0 * inv;
        attnS[r * ASTR + lane + 32] = v1 * inv;
    }
    __syncthreads();

    // ---- softmax over regions per word (lambda=9), emit bf16 attn^T [w][r] ----
    if (threadIdx.x < NW) {
        const int w = threadIdx.x;
        float mx = -1e30f;
        for (int r = 0; r < NR; ++r) mx = fmaxf(mx, attnS[r * ASTR + w]);
        float sum = 0.f;
        for (int r = 0; r < NR; ++r) sum += expf(9.f * (attnS[r * ASTR + w] - mx));
        const float inv = 1.f / sum;
        for (int r = 0; r < NR; ++r)
            attnT[w * TSTR + r] = f2bf(expf(9.f * (attnS[r * ASTR + w] - mx)) * inv);
        for (int r = NR; r < TSTR; ++r) attnT[w * TSTR + r] = 0;   // K padding
    }
    __syncthreads();

    // ---- GEMM2: wei[w][d] = attn^T @ im[i], fused cosine reductions ----
    // wave -> (w-tile = wv>>1, d-half = wv&1); 32 d-tiles each, K = 64 (2 steps)
    const unsigned short* Bt = im_t_bf + (size_t)i * ND * NRP2;
    const float* cap32 = s_f32 + (size_t)j * NW * ND;
    const int wt = wv >> 1;
    const int dh = wv & 1;
    const unsigned short* arow = attnT + (wt * 16 + ln16) * TSTR + sel;
    const v16bf a0 = load_frag(arow);        // K 0..31  (loop-invariant)
    const v16bf a1 = load_frag(arow + 32);   // K 32..63
    float npart[8] = {0.f, 0.f, 0.f, 0.f, 0.f, 0.f, 0.f, 0.f};
    float qpart[8] = {0.f, 0.f, 0.f, 0.f, 0.f, 0.f, 0.f, 0.f};
    for (int dt = dh * 32; dt < dh * 32 + 32; ++dt) {
        v8f acc = {0.f, 0.f, 0.f, 0.f, 0.f, 0.f, 0.f, 0.f};
        const unsigned short* brow = Bt + (size_t)(dt * 16 + ln16) * NRP2 + sel;
        acc = wmma_bf16(a0, load_frag(brow), acc);
        acc = wmma_bf16(a1, load_frag(brow + 32), acc);
        const int d = dt * 16 + ln16;
#pragma unroll
        for (int v = 0; v < 8; ++v) {
            const int w = wt * 16 + v + (hi ? 8 : 0);
            const float wei = acc[v];
            npart[v] += wei * cap32[w * ND + d];
            qpart[v] += wei * wei;
        }
    }
#pragma unroll
    for (int v = 0; v < 8; ++v) {   // reduce across the 16 lanes sharing each row
        float n = npart[v], q = qpart[v];
        for (int m = 1; m <= 8; m <<= 1) {
            n += __shfl_xor(n, m, 32);
            q += __shfl_xor(q, m, 32);
        }
        if (ln16 == 0) {
            const int w = wt * 16 + v + (hi ? 8 : 0);
            numP[dh * NW + w] = n;   // unique (w, d-half) writer: deterministic
            sqP [dh * NW + w] = q;
        }
    }
    __syncthreads();

    // ---- cosine rows + masked LogSumExp (lambda=6) -> scores[i][j] ----
    if (threadIdx.x < NW) {
        const int w = threadIdx.x;
        const float num = numP[w] + numP[NW + w];
        const float nw  = sqrtf(sqP[w] + sqP[NW + w]);
        const float nc  = ncap[(size_t)j * NW + w];
        const float row = num / fmaxf(nc * nw, 1e-8f);
        float e = (w < slen) ? expf(6.f * row) : 0.f;
        for (int m = 16; m; m >>= 1) e += __shfl_xor(e, m, 32);
        if (lane == 0) wred[wv] = e;
    }
    __syncthreads();
    if (threadIdx.x == 0)
        scores[(size_t)i * NB + j] = logf(wred[0] + wred[1]) / 6.f;
}

// ---------------- hinge loss with hardest negatives ----------------

__global__ void k_loss(const float* __restrict__ scores, float* __restrict__ out) {
    __shared__ float diag[NB];
    __shared__ float wsum[4];
    const int t = threadIdx.x;
    diag[t] = scores[(size_t)t * NB + t];
    __syncthreads();
    const float dt_ = diag[t];
    float ms = 0.f, mi = 0.f;
    for (int k = 0; k < NB; ++k) {
        if (k != t) {
            ms = fmaxf(ms, fmaxf(0.f, 0.2f + scores[(size_t)t * NB + k] - dt_));
            mi = fmaxf(mi, fmaxf(0.f, 0.2f + scores[(size_t)k * NB + t] - dt_));
        }
    }
    float tot = ms + mi;
    for (int m = 16; m; m >>= 1) tot += __shfl_xor(tot, m, 32);
    if ((t & 31) == 0) wsum[t >> 5] = tot;
    __syncthreads();
    if (t == 0) out[0] = wsum[0] + wsum[1] + wsum[2] + wsum[3];
}

// ---------------- host launcher ----------------

extern "C" void kernel_launch(void* const* d_in, const int* in_sizes, int n_in,
                              void* d_out, int out_size, void* d_ws, size_t ws_size,
                              hipStream_t stream) {
    const float* im   = (const float*)d_in[0];   // [128][36][1024]
    const float* s    = (const float*)d_in[1];   // [128][64][1024]
    const int*   s_l  = (const int*)d_in[2];     // [128]

    char* ws = (char*)d_ws;
    size_t off = 0;
    unsigned short* im_bf   = (unsigned short*)(ws + off); off += (size_t)NB * NRP * ND * 2;   // 12.6 MB
    unsigned short* im_t_bf = (unsigned short*)(ws + off); off += (size_t)NB * ND * NRP2 * 2;  // 16.8 MB
    unsigned short* s_bf    = (unsigned short*)(ws + off); off += (size_t)NB * NW * ND * 2;    // 16.8 MB
    float*          ncap    = (float*)(ws + off);          off += (size_t)NB * NW * 4;
    float*          scores  = (float*)(ws + off);          off += (size_t)NB * NB * 4;

    k_prep_im<<<NB, 256, 0, stream>>>(im, im_bf, im_t_bf);
    k_prep_s <<<NB, 256, 0, stream>>>(s, s_bf, ncap);
    dim3 grid(NB, NB);
    k_pair<<<grid, 256, 0, stream>>>(im_bf, im_t_bf, s_bf, s, s_l, ncap, scores);
    k_loss<<<1, NB, 0, stream>>>(scores, (float*)d_out);
}
